// GPT2Attention_41420664602947
// MI455X (gfx1250) — compile-verified, matrix operand order checked by reading, not run on hardware
//
#include <hip/hip_runtime.h>

#define EMBED 1024
#define HEADS 16
#define HEAD_DIM 64
#define SEQ 2048
#define BATCH 4

typedef __attribute__((ext_vector_type(16))) __bf16 v16bf;
typedef __attribute__((ext_vector_type(8)))  float  v8f;
typedef __attribute__((ext_vector_type(16))) unsigned short v16u;
typedef __attribute__((ext_vector_type(8)))  unsigned short v8u;
typedef __attribute__((ext_vector_type(4)))  int v4i;

// ---- CDNA5 async global->LDS copy (guarded; falls back to direct global) ----
#if defined(__has_builtin)
#if __has_builtin(__builtin_amdgcn_global_load_async_to_lds_b128)
#define HAVE_ASYNC_LDS 1
#endif
#endif

#ifdef HAVE_ASYNC_LDS
// Builtin signature (from hipcc diagnostics): typed pointers
//   void __builtin_amdgcn_global_load_async_to_lds_b128(
//       v4i addrspace(1)* src, v4i addrspace(3)* dst, imm int off, imm int cpol)
static __device__ __forceinline__ void async_cp16(const unsigned short* src,
                                                  unsigned short* dst) {
  __builtin_amdgcn_global_load_async_to_lds_b128(
      (__attribute__((address_space(1))) v4i*)(src),
      (__attribute__((address_space(3))) v4i*)(dst), 0, 0);
}
#endif
static __device__ __forceinline__ void wait_async0() {
#if defined(__has_builtin) && __has_builtin(__builtin_amdgcn_s_wait_asynccnt)
  __builtin_amdgcn_s_wait_asynccnt(0);
#else
  asm volatile("s_wait_asynccnt 0" ::: "memory");
#endif
}

static __device__ __forceinline__ unsigned short f2bf(float f) {
  union { float f; unsigned u; } v; v.f = f;
  unsigned r = v.u + 0x7FFFu + ((v.u >> 16) & 1u);   // round-to-nearest-even
  return (unsigned short)(r >> 16);
}
static __device__ __forceinline__ v16bf as_bf16(v16u u) {
  union { v16u u; v16bf b; } c; c.u = u; return c.b;
}
static __device__ __forceinline__ v8f zero_v8f() {
  v8f z;
#pragma unroll
  for (int i = 0; i < 8; ++i) z[i] = 0.0f;
  return z;
}
// A/B-operand fragment gather: lane's 16 bf16 = two contiguous 8-element runs
// at p[0..7] and p[16..23]  (K = h8 + (j&7) + (j>=8)*16 pattern)
static __device__ __forceinline__ v16u ld_frag(const unsigned short* p) {
  const v8u lo = *(const v8u*)(p);
  const v8u hi = *(const v8u*)(p + 16);
  v16u u;
#pragma unroll
  for (int e = 0; e < 8; ++e) { u[e] = lo[e]; u[8 + e] = hi[e]; }
  return u;
}

// ---------------------------------------------------------------------------
// One-time: f32 -> bf16 streaming convert (8 elements / thread)
// ---------------------------------------------------------------------------
__global__ __launch_bounds__(256) void convert_f32_bf16(
    const float* __restrict__ src, unsigned short* __restrict__ dst, size_t n)
{
  const size_t i = ((size_t)blockIdx.x * 256 + threadIdx.x) * 8;
  if (i >= n) return;
  const float4 f0 = *(const float4*)(src + i);
  const float4 f1 = *(const float4*)(src + i + 4);
  v8u u;
  u[0]=f2bf(f0.x); u[1]=f2bf(f0.y); u[2]=f2bf(f0.z); u[3]=f2bf(f0.w);
  u[4]=f2bf(f1.x); u[5]=f2bf(f1.y); u[6]=f2bf(f1.z); u[7]=f2bf(f1.w);
  *(v8u*)(dst + i) = u;
}

// ---------------------------------------------------------------------------
// One-time: W[K,N] f32 -> Wt[N,K] bf16 (LDS-tiled 32x32 transpose)
// ---------------------------------------------------------------------------
__global__ __launch_bounds__(256) void transpose_to_bf16(
    const float* __restrict__ W, unsigned short* __restrict__ Wt, int K, int N)
{
  __shared__ float tile[32][33];
  const int tx = threadIdx.x;            // 0..31
  const int ty = threadIdx.y;            // 0..7
  const int nb = blockIdx.x * 32;
  const int kb = blockIdx.y * 32;
#pragma unroll
  for (int r = 0; r < 4; ++r)
    tile[ty + 8 * r][tx] = W[(size_t)(kb + ty + 8 * r) * N + nb + tx];
  __syncthreads();
#pragma unroll
  for (int r = 0; r < 4; ++r)
    Wt[(size_t)(nb + ty + 8 * r) * K + kb + tx] = f2bf(tile[tx][ty + 8 * r]);
}

// ---------------------------------------------------------------------------
// GEMM: out[M,N] = A[M,K](bf16) @ Bt[N,K](bf16)^T + bias, optional col scale.
// Block: 256 thr = 8 waves; tile 128(M) x 64(N) x 32(K).
// A fragments: per-wave direct global b128 loads (rows are unique per wave).
// Bt tile (shared by all 8 waves): double-buffered ASYNC copy into LDS,
// overlapped with WMMA compute, to remove the 8x redundant global traffic.
// ---------------------------------------------------------------------------
template <bool OUT_BF16>
__global__ __launch_bounds__(256) void gemm_bf16_wmma(
    const unsigned short* __restrict__ A,    // [M,K] bf16 row-major
    const unsigned short* __restrict__ Bt,   // [N,K] bf16 (pre-transposed W)
    const float* __restrict__ bias,          // [N]
    void* __restrict__ outv,                 // [M,N] f32 or bf16
    int N, int K, int scale_cols, float scale)
{
  const int tid = threadIdx.x;
  const int lane = tid & 31;
  const int wave = tid >> 5;
  const int rowInFrag = lane & 15;
  const int h8 = (lane >> 4) << 3;

  const int rowBase = blockIdx.y * 128;
  const int colBase = blockIdx.x * 64;

  const unsigned short* Arow = A + (size_t)(rowBase + wave * 16 + rowInFrag) * K;

  v8f acc[4] = { zero_v8f(), zero_v8f(), zero_v8f(), zero_v8f() };

#ifdef HAVE_ASYNC_LDS
  __shared__ unsigned short Bs[2][64][32];   // double-buffered Bt k-tile, 8 KB
  // copy mapping: thread t moves one 16B chunk per buffer fill (256*16B = 4KB)
  const int cRow = tid >> 2;                 // 0..63 (N row within tile)
  const int cCol = (tid & 3) * 8;            // 0,8,16,24 (K elements)
  const unsigned short* bsrc = Bt + (size_t)(colBase + cRow) * K + cCol;
  async_cp16(bsrc, &Bs[0][cRow][cCol]);      // prefill buffer 0 (kb = 0)

  for (int kb = 0; kb < K; kb += 32) {
    const int cur = (kb >> 5) & 1;
    wait_async0();                           // own async stores for 'cur' landed
    __syncthreads();                         // all waves' copies landed / reads done
    if (kb + 32 < K)
      async_cp16(bsrc + kb + 32, &Bs[cur ^ 1][cRow][cCol]);  // overlap next tile
    __builtin_prefetch(Arow + kb + 64, 0, 0);
    const v16bf a = as_bf16(ld_frag(Arow + kb + h8));
#pragma unroll
    for (int n = 0; n < 4; ++n) {
      const v16bf b = as_bf16(ld_frag(&Bs[cur][n * 16 + rowInFrag][h8]));
      acc[n] = __builtin_amdgcn_wmma_f32_16x16x32_bf16(
          false, a, false, b, (short)0, acc[n], false, false);
    }
  }
#else
  const unsigned short* Brow0 = Bt + (size_t)(colBase + rowInFrag) * K;
  const size_t bStep = (size_t)16 * K;
  for (int kb = 0; kb < K; kb += 32) {
    __builtin_prefetch(Arow + kb + 64, 0, 0);
    const v16bf a = as_bf16(ld_frag(Arow + kb + h8));
#pragma unroll
    for (int n = 0; n < 4; ++n) {
      const v16bf b = as_bf16(ld_frag(Brow0 + n * bStep + kb + h8));
      acc[n] = __builtin_amdgcn_wmma_f32_16x16x32_bf16(
          false, a, false, b, (short)0, acc[n], false, false);
    }
  }
#endif

  // C/D layout: lane holds rows M=(lane/16)*8+r, col N=lane%16
  const int outRow0 = rowBase + wave * 16 + h8;
  const int nCol = lane & 15;
#pragma unroll
  for (int n = 0; n < 4; ++n) {
    const int col = colBase + n * 16 + nCol;
    const float sc = (col < scale_cols) ? scale : 1.0f;
    const float bv = bias[col];
#pragma unroll
    for (int r = 0; r < 8; ++r) {
      const float v = (acc[n][r] + bv) * sc;
      if (OUT_BF16) ((unsigned short*)outv)[(size_t)(outRow0 + r) * N + col] = f2bf(v);
      else          ((float*)outv)[(size_t)(outRow0 + r) * N + col] = v;
    }
  }
}

// ---------------------------------------------------------------------------
// Flash attention over bf16 qkv[B*S, 3E] (Q pre-scaled by 1/sqrt(D)).
// Block: 128 thr = 4 waves; wave w owns query rows [qb+16w, qb+16w+16).
// K tile (shared by all 4 waves) async-staged into LDS; V transposed into LDS;
// P bounced through per-wave LDS for the C-layout -> A-layout relayout.
// ---------------------------------------------------------------------------
__global__ __launch_bounds__(128) void flash_attn_kernel(
    const unsigned short* __restrict__ qkv, unsigned short* __restrict__ out)
{
#ifdef HAVE_ASYNC_LDS
  __shared__ unsigned short Ks[32][64];       // K rows (fragment-native), 4 KB
#endif
  __shared__ unsigned short Vt[64][32];       // V^T [d][key] bf16, 4 KB
  __shared__ unsigned short Plds[4][16][32];  // per-wave P bounce, 4 KB

  const int tid = threadIdx.x;
  const int lane = tid & 31;
  const int wave = tid >> 5;
  const int rowInFrag = lane & 15;
  const int h8 = (lane >> 4) << 3;

  const int qb = blockIdx.x * 64;
  const int b  = blockIdx.y / HEADS;
  const int h  = blockIdx.y % HEADS;
  const size_t rowStride = 3 * EMBED;

  // Q fragments (already scaled by 1/8 in the QKV epilogue)
  v16bf qf[2];
  {
    const unsigned short* qp =
        qkv + (size_t)(b * SEQ + qb + wave * 16 + rowInFrag) * rowStride + h * HEAD_DIM;
    qf[0] = as_bf16(ld_frag(qp + h8));
    qf[1] = as_bf16(ld_frag(qp + 32 + h8));
  }

  float mrow[8], lrow[8];
  v8f o[4] = { zero_v8f(), zero_v8f(), zero_v8f(), zero_v8f() };
#pragma unroll
  for (int r = 0; r < 8; ++r) { mrow[r] = -3.0e30f; lrow[r] = 0.0f; }

  const int kend = qb + 64;                   // causal upper bound for this block
  for (int kb = 0; kb < kend; kb += 32) {
#ifdef HAVE_ASYNC_LDS
    // ---- async-stage K tile rows: thread t -> (key = t>>2, 16-elem d chunk) ----
    {
      const int key   = tid >> 2;
      const int dpart = (tid & 3) * 16;
      const unsigned short* ksrc =
          qkv + (size_t)(b * SEQ + kb + key) * rowStride + EMBED + h * HEAD_DIM + dpart;
      async_cp16(ksrc,     &Ks[key][dpart]);
      async_cp16(ksrc + 8, &Ks[key][dpart + 8]);
    }
#endif
    // ---- stage V transposed: thread -> (key = tid&31, d-run = (tid>>5)*16) ----
    {
      const unsigned short* vp =
          qkv + (size_t)(b * SEQ + kb + (tid & 31)) * rowStride
          + 2 * EMBED + h * HEAD_DIM + (tid >> 5) * 16;
      const v8u a0 = *(const v8u*)(vp);
      const v8u a1 = *(const v8u*)(vp + 8);
      const int drun = (tid >> 5) * 16;
#pragma unroll
      for (int e = 0; e < 8; ++e) {
        Vt[drun + e][tid & 31]     = a0[e];
        Vt[drun + 8 + e][tid & 31] = a1[e];
      }
    }
#ifdef HAVE_ASYNC_LDS
    wait_async0();
#endif
    __syncthreads();

    // ---- S = Q @ K^T ----
    v8f s0 = zero_v8f(), s1 = zero_v8f();
    {
#ifdef HAVE_ASYNC_LDS
      const unsigned short* kp0 = &Ks[rowInFrag][0];
      const unsigned short* kp1 = &Ks[16 + rowInFrag][0];
#else
      const unsigned short* kp0 =
          qkv + (size_t)(b * SEQ + kb + rowInFrag) * rowStride + EMBED + h * HEAD_DIM;
      const unsigned short* kp1 = kp0 + 16 * rowStride;
#endif
#pragma unroll
      for (int c = 0; c < 2; ++c) {
        const v16bf k0 = as_bf16(ld_frag(kp0 + c * 32 + h8));
        const v16bf k1 = as_bf16(ld_frag(kp1 + c * 32 + h8));
        s0 = __builtin_amdgcn_wmma_f32_16x16x32_bf16(
            false, qf[c], false, k0, (short)0, s0, false, false);
        s1 = __builtin_amdgcn_wmma_f32_16x16x32_bf16(
            false, qf[c], false, k1, (short)0, s1, false, false);
      }
    }
    // ---- causal mask (reference: masked bias -10000, post-scale) ----
    const int key0 = kb + (lane & 15);
#pragma unroll
    for (int r = 0; r < 8; ++r) {
      const int qr = qb + wave * 16 + h8 + r;
      if (key0 > qr)      s0[r] = -10000.0f;
      if (key0 + 16 > qr) s1[r] = -10000.0f;
    }
    // ---- online softmax (row stats live within 16-lane halves) ----
    float p0[8], p1[8];
#pragma unroll
    for (int r = 0; r < 8; ++r) {
      float rm = fmaxf(s0[r], s1[r]);
      rm = fmaxf(rm, __shfl_xor(rm, 1, 32));
      rm = fmaxf(rm, __shfl_xor(rm, 2, 32));
      rm = fmaxf(rm, __shfl_xor(rm, 4, 32));
      rm = fmaxf(rm, __shfl_xor(rm, 8, 32));
      const float mnew  = fmaxf(mrow[r], rm);
      const float alpha = __expf(mrow[r] - mnew);
      mrow[r] = mnew;
      p0[r] = __expf(s0[r] - mnew);
      p1[r] = __expf(s1[r] - mnew);
      float rs = p0[r] + p1[r];
      rs += __shfl_xor(rs, 1, 32);
      rs += __shfl_xor(rs, 2, 32);
      rs += __shfl_xor(rs, 4, 32);
      rs += __shfl_xor(rs, 8, 32);
      lrow[r] = lrow[r] * alpha + rs;
#pragma unroll
      for (int n = 0; n < 4; ++n) o[n][r] = o[n][r] * alpha;
    }
    // ---- bounce P through per-wave LDS: C-layout -> A-layout bf16 ----
#pragma unroll
    for (int r = 0; r < 8; ++r) {
      const int M = h8 + r;
      Plds[wave][M][lane & 15]        = f2bf(p0[r]);
      Plds[wave][M][16 + (lane & 15)] = f2bf(p1[r]);
    }
    // same-wave DS ops are in-order; no workgroup barrier needed here
    v16u pu;
    {
      const unsigned short* pr = &Plds[wave][rowInFrag][0];
#pragma unroll
      for (int e = 0; e < 8; ++e) {
        pu[e]     = pr[h8 + e];
        pu[8 + e] = pr[h8 + 16 + e];
      }
    }
    const v16bf pf = as_bf16(pu);
    // ---- O += P @ V ----
#pragma unroll
    for (int n = 0; n < 4; ++n) {
      const int d = n * 16 + rowInFrag;
      v16u vu;
      const unsigned short* vr = &Vt[d][0];
#pragma unroll
      for (int e = 0; e < 8; ++e) {
        vu[e]     = vr[h8 + e];
        vu[8 + e] = vr[h8 + 16 + e];
      }
      o[n] = __builtin_amdgcn_wmma_f32_16x16x32_bf16(
          false, pf, false, as_bf16(vu), (short)0, o[n], false, false);
    }
    __syncthreads();   // Ks/Vt re-staged next iteration
  }

  // ---- normalize, write attn_out as bf16 [B,S,E] at column h*64 ----
#pragma unroll
  for (int r = 0; r < 8; ++r) {
    const float inv = 1.0f / lrow[r];
    const int qr = qb + wave * 16 + h8 + r;
    unsigned short* op = out + (size_t)(b * SEQ + qr) * EMBED + h * HEAD_DIM;
#pragma unroll
    for (int n = 0; n < 4; ++n) {
      op[n * 16 + (lane & 15)] = f2bf(o[n][r] * inv);
    }
  }
}

// ---------------------------------------------------------------------------
extern "C" void kernel_launch(void* const* d_in, const int* in_sizes, int n_in,
                              void* d_out, int out_size, void* d_ws, size_t ws_size,
                              hipStream_t stream) {
  const float* hidden   = (const float*)d_in[0];  // [4,2048,1024] f32
  const float* c_attn_w = (const float*)d_in[1];  // [1024,3072]
  const float* c_attn_b = (const float*)d_in[2];  // [3072]
  const float* c_proj_w = (const float*)d_in[3];  // [1024,1024]
  const float* c_proj_b = (const float*)d_in[4];  // [1024]
  float* out = (float*)d_out;                     // [4,2048,1024] f32

  const int M = BATCH * SEQ;                      // 8192
  unsigned short* ws = (unsigned short*)d_ws;
  unsigned short* Xbf    = ws;                                   // 8192x1024  16 MB
  unsigned short* Wqkvt  = Xbf    + (size_t)M * EMBED;           // 3072x1024   6 MB
  unsigned short* Wprojt = Wqkvt  + (size_t)3 * EMBED * EMBED;   // 1024x1024   2 MB
  unsigned short* qkvbf  = Wprojt + (size_t)EMBED * EMBED;       // 8192x3072  48 MB
  unsigned short* attnbf = qkvbf  + (size_t)M * 3 * EMBED;       // 8192x1024  16 MB

  // 0) One-time bf16 conversion / weight transposes
  {
    const size_t nX = (size_t)M * EMBED;
    convert_f32_bf16<<<dim3((unsigned)(nX / (8 * 256))), 256, 0, stream>>>(hidden, Xbf, nX);
    transpose_to_bf16<<<dim3(3 * EMBED / 32, EMBED / 32), dim3(32, 8), 0, stream>>>(
        c_attn_w, Wqkvt, EMBED, 3 * EMBED);
    transpose_to_bf16<<<dim3(EMBED / 32, EMBED / 32), dim3(32, 8), 0, stream>>>(
        c_proj_w, Wprojt, EMBED, EMBED);
  }
  // 1) QKV projection -> bf16, Q columns (col < E) pre-scaled by 1/sqrt(64)
  gemm_bf16_wmma<true><<<dim3(3 * EMBED / 64, M / 128), 256, 0, stream>>>(
      Xbf, Wqkvt, c_attn_b, qkvbf, 3 * EMBED, EMBED, EMBED, 0.125f);
  // 2) Causal flash attention per (b, h, 64-row query block)
  flash_attn_kernel<<<dim3(SEQ / 64, BATCH * HEADS), 128, 0, stream>>>(qkvbf, attnbf);
  // 3) Output projection -> f32 result
  gemm_bf16_wmma<false><<<dim3(EMBED / 64, M / 128), 256, 0, stream>>>(
      attnbf, Wprojt, c_proj_b, out, EMBED, EMBED, 0, 1.0f);
}
